// LocalUTrans_58317065945988
// MI455X (gfx1250) — compile-verified, matrix operand order checked by reading, not run on hardware
//
#include <hip/hip_runtime.h>
#include <hip/hip_bf16.h>
#include <math.h>

// ---------------------------------------------------------------------------
// Types for WMMA (gfx1250, wave32)
// ---------------------------------------------------------------------------
typedef __attribute__((ext_vector_type(16))) __bf16          v16bf;
typedef __attribute__((ext_vector_type(8)))  float           v8f;
typedef __attribute__((ext_vector_type(8)))  unsigned short  v8us;

__device__ __forceinline__ unsigned short f2bf(float f) {
  union { float f; unsigned u; } x; x.f = f;
  unsigned u = x.u;
  if ((u & 0x7fffffffu) > 0x7f800000u) return (unsigned short)((u >> 16) | 0x40); // NaN
  unsigned r = u + 0x7fffu + ((u >> 16) & 1u); // round-to-nearest-even
  return (unsigned short)(r >> 16);
}

// ---------------------------------------------------------------------------
// WMMA bf16 GEMM: C = alpha * A @ op(B) (+bias)(+act)(+res)
//   A: M x K row-major (lda, K-contiguous), bf16 bits in ushort
//   BNT=true  -> B is N x K row-major (NT, K-contiguous)   [weights / K-matrix]
//   BNT=false -> B is K x N row-major (NN, strided loads)  [V matrix]
//   store_t=1 -> C[n*ldc+m] (CHW conv output), else C[m*ldc+n]
//   M mult of 16; N mult of 32; K mult of 32; lda/ldb mult of 8 (16B aligned)
// One wave computes a 16x32 output tile (two 16x16 WMMA tiles sharing A).
// ---------------------------------------------------------------------------
template <bool BNT>
__global__ __launch_bounds__(32)
void gemm_bf16_k(const unsigned short* __restrict__ A, int lda,
                 const unsigned short* __restrict__ B, int ldb,
                 float* __restrict__ C, int ldc, int store_t,
                 int M, int N, int K, float alpha,
                 const float* __restrict__ bias, int act,
                 const float* __restrict__ res, int ldres)
{
  const int n0   = blockIdx.x * 32;
  const int m0   = blockIdx.y * 16;
  const int lane = threadIdx.x;
  const int row  = lane & 15;
  const int kb   = (lane >> 4) * 8;

  v8f acc0 = {0.f, 0.f, 0.f, 0.f, 0.f, 0.f, 0.f, 0.f};
  v8f acc1 = {0.f, 0.f, 0.f, 0.f, 0.f, 0.f, 0.f, 0.f};
  union Frag { v8us h[2]; unsigned short u[16]; v16bf v; };

  const unsigned short* pa = A + (size_t)(m0 + row) * lda + kb;
  const unsigned short* pb0;
  const unsigned short* pb1;
  if (BNT) {
    pb0 = B + (size_t)(n0 + row) * ldb + kb;
    pb1 = pb0 + (size_t)16 * ldb;
  } else {
    pb0 = B + (size_t)kb * ldb + (n0 + row);
    pb1 = pb0 + 16;
  }

  for (int k0 = 0; k0 < K; k0 += 32) {
    Frag fa, fb0, fb1;
    fa.h[0] = ((const v8us*)pa)[0];
    fa.h[1] = ((const v8us*)pa)[2];            // +16 elements
    __builtin_prefetch(pa + 64, 0, 1);         // global_prefetch_b8
    if (BNT) {
      fb0.h[0] = ((const v8us*)pb0)[0];
      fb0.h[1] = ((const v8us*)pb0)[2];
      fb1.h[0] = ((const v8us*)pb1)[0];
      fb1.h[1] = ((const v8us*)pb1)[2];
      __builtin_prefetch(pb0 + 64, 0, 1);
      pb0 += 32;
      pb1 += 32;
    } else {
#pragma unroll
      for (int j = 0; j < 8; ++j) {
        fb0.u[j]     = pb0[(size_t)j * ldb];
        fb0.u[8 + j] = pb0[(size_t)(16 + j) * ldb];
        fb1.u[j]     = pb1[(size_t)j * ldb];
        fb1.u[8 + j] = pb1[(size_t)(16 + j) * ldb];
      }
      pb0 += (size_t)32 * ldb;
      pb1 += (size_t)32 * ldb;
    }
    pa += 32;
    acc0 = __builtin_amdgcn_wmma_f32_16x16x32_bf16(false, fa.v, false, fb0.v,
                                                   (short)0, acc0, false, false);
    acc1 = __builtin_amdgcn_wmma_f32_16x16x32_bf16(false, fa.v, false, fb1.v,
                                                   (short)0, acc1, false, false);
  }

  const int nloc  = lane & 15;
  const int mbase = (lane >> 4) * 8;
#pragma unroll
  for (int t = 0; t < 2; ++t) {
    v8f av = (t == 0) ? acc0 : acc1;
#pragma unroll
    for (int r = 0; r < 8; ++r) {
      int m = m0 + mbase + r;
      int n = n0 + t * 16 + nloc;
      float v = alpha * av[r];
      if (bias) v += bias[n];
      if (act == 1) v = 0.5f * v * (1.0f + erff(v * 0.70710678118654752f)); // exact GELU
      if (res) v += res[(size_t)m * ldres + n];
      if (store_t) C[(size_t)n * ldc + m] = v;
      else         C[(size_t)m * ldc + n] = v;
    }
  }
}

// ---------------------------------------------------------------------------
// im2col (3x3, SAME, single image NCHW) -> (HW x Kpad) bf16, zero-padded K
// ---------------------------------------------------------------------------
__global__ void im2col_bf16(const float* __restrict__ in, unsigned short* __restrict__ out,
                            int H, int W, int Cin, int Kpad)
{
  size_t idx = (size_t)blockIdx.x * blockDim.x + threadIdx.x;
  size_t total = (size_t)H * W * Kpad;
  if (idx >= total) return;
  int k = (int)(idx % Kpad);
  size_t m = idx / Kpad;
  float v = 0.f;
  if (k < Cin * 9) {
    int ci = k / 9, r = k % 9;
    int y = (int)(m / W) + r / 3 - 1;
    int x = (int)(m % W) + r % 3 - 1;
    if (y >= 0 && y < H && x >= 0 && x < W)
      v = in[((size_t)ci * H + y) * W + x];
  }
  out[idx] = f2bf(v);
}

// weights (N, K) f32 -> (N, Kpad) bf16, zero-padded
__global__ void stage_w_bf16(const float* __restrict__ w, unsigned short* __restrict__ out,
                             int N, int K, int Kpad)
{
  size_t idx = (size_t)blockIdx.x * blockDim.x + threadIdx.x;
  if (idx >= (size_t)N * Kpad) return;
  int k = (int)(idx % Kpad);
  size_t n = idx / Kpad;
  out[idx] = f2bf(k < K ? w[n * K + k] : 0.f);
}

__global__ void cvt_bf16(const float* __restrict__ in, unsigned short* __restrict__ out, size_t n)
{
  size_t i = (size_t)blockIdx.x * blockDim.x + threadIdx.x;
  if (i < n) out[i] = f2bf(in[i]);
}

// ---------------------------------------------------------------------------
// BatchNorm (training mode, batch stats) + ReLU
// ---------------------------------------------------------------------------
__global__ __launch_bounds__(256)
void bn_stats(const float* __restrict__ x, int Nimg, int C, int HW,
              float* __restrict__ mean, float* __restrict__ istd)
{
  int c = blockIdx.x;
  __shared__ float ss[256], sq[256];
  float s = 0.f, q = 0.f;
  int tot = Nimg * HW;
  for (int i = threadIdx.x; i < tot; i += 256) {
    int n = i / HW, p = i % HW;
    float v = x[((size_t)n * C + c) * HW + p];
    s += v; q += v * v;
  }
  ss[threadIdx.x] = s; sq[threadIdx.x] = q;
  __syncthreads();
  for (int st = 128; st > 0; st >>= 1) {
    if (threadIdx.x < st) { ss[threadIdx.x] += ss[threadIdx.x + st]; sq[threadIdx.x] += sq[threadIdx.x + st]; }
    __syncthreads();
  }
  if (threadIdx.x == 0) {
    float m = ss[0] / tot;
    float var = sq[0] / tot - m * m;
    mean[c] = m;
    istd[c] = rsqrtf(var + 1e-5f);
  }
}

__global__ void bn_relu(const float* __restrict__ x, float* __restrict__ out,
                        const float* __restrict__ g, const float* __restrict__ be,
                        const float* __restrict__ mean, const float* __restrict__ istd,
                        int C, int HW, size_t total)
{
  size_t i = (size_t)blockIdx.x * blockDim.x + threadIdx.x;
  if (i >= total) return;
  int c = (int)((i / HW) % C);
  float v = (x[i] - mean[c]) * istd[c] * g[c] + be[c];
  out[i] = v > 0.f ? v : 0.f;
}

__global__ void maxpool2(const float* __restrict__ in, float* __restrict__ out,
                         int H, int W, size_t total)
{
  size_t i = (size_t)blockIdx.x * blockDim.x + threadIdx.x;
  if (i >= total) return;
  int ow = W >> 1, oh = H >> 1;
  int x = (int)(i % ow);
  int y = (int)((i / ow) % oh);
  size_t nc = i / ((size_t)ow * oh);
  const float* p = in + (nc * H + 2 * y) * W + 2 * x;
  float a = fmaxf(p[0], p[1]);
  float b = fmaxf(p[W], p[W + 1]);
  out[i] = fmaxf(a, b);
}

// bilinear x2, align_corners=True
__global__ void upsample2x(const float* __restrict__ in, float* __restrict__ out,
                           int H, int W, size_t total)
{
  size_t i = (size_t)blockIdx.x * blockDim.x + threadIdx.x;
  if (i >= total) return;
  int OW = 2 * W, OH = 2 * H;
  int ox = (int)(i % OW);
  int oy = (int)((i / OW) % OH);
  size_t nc = i / ((size_t)OW * OH);
  float yc = oy * (float)(H - 1) / (float)(OH - 1);
  float xc = ox * (float)(W - 1) / (float)(OW - 1);
  int y0 = (int)floorf(yc); int y1 = min(y0 + 1, H - 1); float wy = yc - y0;
  int x0 = (int)floorf(xc); int x1 = min(x0 + 1, W - 1); float wx = xc - x0;
  const float* p = in + nc * (size_t)H * W;
  float r0 = p[(size_t)y0 * W + x0] * (1.f - wy) + p[(size_t)y1 * W + x0] * wy;
  float r1 = p[(size_t)y0 * W + x1] * (1.f - wy) + p[(size_t)y1 * W + x1] * wy;
  out[i] = r0 * (1.f - wx) + r1 * wx;
}

__global__ void copy_ch(const float* __restrict__ src, float* __restrict__ dst,
                        int Cs, int Ctot, int choff, int HW, size_t total)
{
  size_t i = (size_t)blockIdx.x * blockDim.x + threadIdx.x;
  if (i >= total) return;
  int p = (int)(i % HW);
  int c = (int)((i / HW) % Cs);
  size_t n = i / ((size_t)HW * Cs);
  dst[((size_t)n * Ctot + choff + c) * HW + p] = src[i];
}

__global__ void add_pe_copy(float* __restrict__ x, const float* __restrict__ pe,
                            float* __restrict__ xres, size_t chw, size_t total)
{
  size_t i = (size_t)blockIdx.x * blockDim.x + threadIdx.x;
  if (i >= total) return;
  float v = x[i] + pe[i % chw];
  x[i] = v; xres[i] = v;
}

__global__ void add_inplace(float* __restrict__ x, const float* __restrict__ y, size_t n)
{
  size_t i = (size_t)blockIdx.x * blockDim.x + threadIdx.x;
  if (i < n) x[i] += y[i];
}

__global__ __launch_bounds__(256)
void layernorm_k(const float* __restrict__ x, float* __restrict__ out,
                 const float* __restrict__ g, const float* __restrict__ b, int E)
{
  size_t row = blockIdx.x;
  const float* xr = x + row * E;
  float* o = out + row * E;
  __shared__ float ss[256], sq[256];
  float s = 0.f, q = 0.f;
  for (int i = threadIdx.x; i < E; i += 256) { float v = xr[i]; s += v; q += v * v; }
  ss[threadIdx.x] = s; sq[threadIdx.x] = q;
  __syncthreads();
  for (int st = 128; st > 0; st >>= 1) {
    if (threadIdx.x < st) { ss[threadIdx.x] += ss[threadIdx.x + st]; sq[threadIdx.x] += sq[threadIdx.x + st]; }
    __syncthreads();
  }
  float m = ss[0] / E;
  float inv = rsqrtf(sq[0] / E - m * m + 1e-5f);
  for (int i = threadIdx.x; i < E; i += 256)
    o[i] = (xr[i] - m) * inv * g[i] + b[i];
}

// row-wise softmax over L, output bf16 probs
__global__ __launch_bounds__(256)
void softmax_bf16(const float* __restrict__ x, unsigned short* __restrict__ out, int L)
{
  size_t row = blockIdx.x;
  const float* xr = x + row * (size_t)L;
  __shared__ float sh[256];
  float mx = -3.4e38f;
  for (int i = threadIdx.x; i < L; i += 256) mx = fmaxf(mx, xr[i]);
  sh[threadIdx.x] = mx; __syncthreads();
  for (int st = 128; st > 0; st >>= 1) {
    if (threadIdx.x < st) sh[threadIdx.x] = fmaxf(sh[threadIdx.x], sh[threadIdx.x + st]);
    __syncthreads();
  }
  mx = sh[0]; __syncthreads();
  float s = 0.f;
  for (int i = threadIdx.x; i < L; i += 256) s += expf(xr[i] - mx);
  sh[threadIdx.x] = s; __syncthreads();
  for (int st = 128; st > 0; st >>= 1) {
    if (threadIdx.x < st) sh[threadIdx.x] += sh[threadIdx.x + st];
    __syncthreads();
  }
  float inv = 1.f / sh[0];
  for (int i = threadIdx.x; i < L; i += 256)
    out[row * (size_t)L + i] = f2bf(expf(xr[i] - mx) * inv);
}

__global__ void conv1x1_64_1(const float* __restrict__ in, const float* __restrict__ w,
                             const float* __restrict__ b, float* __restrict__ out,
                             int HW, size_t total)
{
  size_t i = (size_t)blockIdx.x * blockDim.x + threadIdx.x;
  if (i >= total) return;
  int p = (int)(i % HW);
  size_t n = i / HW;
  const float* pp = in + n * (size_t)64 * HW + p;
  float s = b[0];
#pragma unroll
  for (int c = 0; c < 64; ++c) s += pp[(size_t)c * HW] * w[c];
  out[i] = s;
}

// ---------------------------------------------------------------------------
// Host orchestration
// ---------------------------------------------------------------------------
static inline unsigned cdiv(size_t a, size_t b) { return (unsigned)((a + b - 1) / b); }

static void conv3x3(hipStream_t st, const float* in, int Nimg, int Cin, int H, int W,
                    const float* w, const float* bias, int Cout,
                    float* out, int choff, int ctot,
                    unsigned short* sim, unsigned short* sw)
{
  int K = Cin * 9, Kpad = (K + 31) & ~31, HW = H * W;
  size_t wtot = (size_t)Cout * Kpad;
  stage_w_bf16<<<cdiv(wtot, 256), 256, 0, st>>>(w, sw, Cout, K, Kpad);
  for (int img = 0; img < Nimg; ++img) {
    size_t itot = (size_t)HW * Kpad;
    im2col_bf16<<<cdiv(itot, 256), 256, 0, st>>>(in + (size_t)img * Cin * HW, sim, H, W, Cin, Kpad);
    dim3 g(Cout / 32, HW / 16);
    gemm_bf16_k<true><<<g, 32, 0, st>>>(sim, Kpad, sw, Kpad,
                                        out + ((size_t)img * ctot + choff) * HW, HW, 1,
                                        HW, Cout, Kpad, 1.f, bias, 0, nullptr, 0);
  }
}

static void cbr(hipStream_t st, const float* in, int Nimg, int Cin, int H, int W,
                const float* w, const float* b, const float* g, const float* be,
                int Cout, float* pre, float* out, float* mean, float* istd,
                unsigned short* sim, unsigned short* sw)
{
  conv3x3(st, in, Nimg, Cin, H, W, w, b, Cout, pre, 0, Cout, sim, sw);
  int HW = H * W;
  bn_stats<<<Cout, 256, 0, st>>>(pre, Nimg, Cout, HW, mean, istd);
  size_t tot = (size_t)Nimg * Cout * HW;
  bn_relu<<<cdiv(tot, 256), 256, 0, st>>>(pre, out, g, be, mean, istd, Cout, HW, tot);
}

extern "C" void kernel_launch(void* const* d_in, const int* in_sizes, int n_in,
                              void* d_out, int out_size, void* d_ws, size_t ws_size,
                              hipStream_t stream)
{
  (void)in_sizes; (void)n_in; (void)out_size; (void)ws_size;
  char* ws = (char*)d_ws;
  size_t off = 0;
  auto alloc = [&](size_t bytes) -> void* {
    void* p = (void*)(ws + off);
    off += (bytes + 255) & ~(size_t)255;
    return p;
  };

  float* T1 = (float*)alloc(37748736ull * 4);  // ping (pre-BN / upsample)
  float* T2 = (float*)alloc(18874368ull * 4);  // pong (activations / seq tensor)
  float* CC = (float*)alloc(37748736ull * 4);  // concat buffer / transformer temps
  float* S1 = (float*)alloc(18874368ull * 4);  // skip 64ch @192^2
  float* S2 = (float*)alloc(9437184ull * 4);   // skip 128ch @96^2
  float* S3 = (float*)alloc(4718592ull * 4);   // skip 256ch @48^2
  float* XR = (float*)alloc(2359296ull * 4);   // transformer outer residual
  float* LN = (float*)alloc(2359296ull * 4);   // layernorm output
  float* VA = (float*)alloc(2359296ull * 4);   // attention values (B,S,E)
  float* MEAN = (float*)alloc(1024 * 4);
  float* ISTD = (float*)alloc(1024 * 4);
  unsigned short* SIM = (unsigned short*)alloc(42467328ull * 2); // im2col / act / probs bf16
  unsigned short* SQK = (unsigned short*)alloc(7077888ull * 2);  // qkv bf16
  unsigned short* SW  = (unsigned short*)alloc(2621440ull * 2);  // weights bf16

  // transformer f32 temporaries alias into CC (dead during transformer)
  float* QKV = CC;                 // 8*576*1536 = 7,077,888
  float* ATT = CC + 7077888;       // 64*576*576 = 21,233,664
  float* F1  = CC + 28311552;      // 8*576*2048 = 9,437,184

  auto P = [&](int i) { return (const float*)d_in[i]; };
  const float* x = P(0);

  // ------------------------- Encoder -------------------------
  // level 0: 1 -> 64 @ 192x192
  cbr(stream, x,  8,  1, 192, 192, P(1), P(2), P(3), P(4), 64, T1, T2, MEAN, ISTD, SIM, SW);
  cbr(stream, T2, 8, 64, 192, 192, P(5), P(6), P(7), P(8), 64, T1, S1, MEAN, ISTD, SIM, SW);
  { size_t t = 8ull * 64 * 96 * 96;
    maxpool2<<<cdiv(t, 256), 256, 0, stream>>>(S1, T2, 192, 192, t); }
  // level 1: 64 -> 128 @ 96x96
  cbr(stream, T2, 8,  64, 96, 96, P(9),  P(10), P(11), P(12), 128, T1, T2, MEAN, ISTD, SIM, SW);
  cbr(stream, T2, 8, 128, 96, 96, P(13), P(14), P(15), P(16), 128, T1, S2, MEAN, ISTD, SIM, SW);
  { size_t t = 8ull * 128 * 48 * 48;
    maxpool2<<<cdiv(t, 256), 256, 0, stream>>>(S2, T2, 96, 96, t); }
  // level 2: 128 -> 256 @ 48x48
  cbr(stream, T2, 8, 128, 48, 48, P(17), P(18), P(19), P(20), 256, T1, T2, MEAN, ISTD, SIM, SW);
  cbr(stream, T2, 8, 256, 48, 48, P(21), P(22), P(23), P(24), 256, T1, S3, MEAN, ISTD, SIM, SW);
  { size_t t = 8ull * 256 * 24 * 24;
    maxpool2<<<cdiv(t, 256), 256, 0, stream>>>(S3, T2, 48, 48, t); }
  // level 3: 256 -> 512 @ 24x24 (no pool)
  cbr(stream, T2, 8, 256, 24, 24, P(25), P(26), P(27), P(28), 512, T1, T2, MEAN, ISTD, SIM, SW);
  cbr(stream, T2, 8, 512, 24, 24, P(29), P(30), P(31), P(32), 512, T1, T2, MEAN, ISTD, SIM, SW);

  // x = x + pe ; keep outer residual
  { size_t t = 2359296ull;
    add_pe_copy<<<cdiv(t, 256), 256, 0, stream>>>(T2, P(33), XR, 294912ull, t); }

  // ------------------------- Transformer (x2) -------------------------
  const int S = 576, E = 512, NH = 8, HD = 64, Mf = 8 * S;
  for (int blk = 0; blk < 2; ++blk) {
    int bse = 34 + blk * 12;
    const float *ln1g = P(bse + 0), *ln1b = P(bse + 1);
    const float *qw = P(bse + 2), *qb = P(bse + 3);
    const float *ow = P(bse + 4), *ob = P(bse + 5);
    const float *ln2g = P(bse + 6), *ln2b = P(bse + 7);
    const float *f1w = P(bse + 8), *f1b = P(bse + 9);
    const float *f2w = P(bse + 10), *f2b = P(bse + 11);

    // --- attention ---
    layernorm_k<<<Mf, 256, 0, stream>>>(T2, LN, ln1g, ln1b, E);
    cvt_bf16<<<cdiv((size_t)Mf * E, 256), 256, 0, stream>>>(LN, SIM, (size_t)Mf * E);
    cvt_bf16<<<cdiv(1536ull * 512, 256), 256, 0, stream>>>(qw, SW, 1536ull * 512);
    gemm_bf16_k<true><<<dim3(1536 / 32, Mf / 16), 32, 0, stream>>>(
        SIM, 512, SW, 512, QKV, 1536, 0, Mf, 1536, 512, 1.f, qb, 0, nullptr, 0);
    cvt_bf16<<<cdiv((size_t)Mf * 1536, 256), 256, 0, stream>>>(QKV, SQK, (size_t)Mf * 1536);

    for (int b = 0; b < 8; ++b)
      for (int h = 0; h < NH; ++h) {
        const unsigned short* Q = SQK + (size_t)b * S * 1536 + h * 192;
        const unsigned short* Kk = Q + 64;
        float* L = ATT + (size_t)(b * NH + h) * S * S;
        gemm_bf16_k<true><<<dim3(S / 32, S / 16), 32, 0, stream>>>(
            Q, 1536, Kk, 1536, L, S, 0, S, S, 64, 0.125f, nullptr, 0, nullptr, 0);
      }
    softmax_bf16<<<8 * NH * S, 256, 0, stream>>>(ATT, SIM, S);
    for (int b = 0; b < 8; ++b)
      for (int h = 0; h < NH; ++h) {
        const unsigned short* Pm = SIM + (size_t)(b * NH + h) * S * S;
        const unsigned short* V = SQK + (size_t)b * S * 1536 + h * 192 + 128;
        float* O = VA + (size_t)b * S * E + h * HD;
        gemm_bf16_k<false><<<dim3(HD / 32, S / 16), 32, 0, stream>>>(
            Pm, S, V, 1536, O, E, 0, S, HD, S, 1.f, nullptr, 0, nullptr, 0);
      }
    // output projection + residual (in place on T2)
    cvt_bf16<<<cdiv((size_t)Mf * E, 256), 256, 0, stream>>>(VA, SIM, (size_t)Mf * E);
    cvt_bf16<<<cdiv(512ull * 512, 256), 256, 0, stream>>>(ow, SW, 512ull * 512);
    gemm_bf16_k<true><<<dim3(E / 32, Mf / 16), 32, 0, stream>>>(
        SIM, 512, SW, 512, T2, E, 0, Mf, E, 512, 1.f, ob, 0, T2, E);

    // --- MLP ---
    layernorm_k<<<Mf, 256, 0, stream>>>(T2, LN, ln2g, ln2b, E);
    cvt_bf16<<<cdiv((size_t)Mf * E, 256), 256, 0, stream>>>(LN, SIM, (size_t)Mf * E);
    cvt_bf16<<<cdiv(2048ull * 512, 256), 256, 0, stream>>>(f1w, SW, 2048ull * 512);
    gemm_bf16_k<true><<<dim3(2048 / 32, Mf / 16), 32, 0, stream>>>(
        SIM, 512, SW, 512, F1, 2048, 0, Mf, 2048, 512, 1.f, f1b, 1 /*gelu*/, nullptr, 0);
    cvt_bf16<<<cdiv((size_t)Mf * 2048, 256), 256, 0, stream>>>(F1, SIM, (size_t)Mf * 2048);
    cvt_bf16<<<cdiv(512ull * 2048, 256), 256, 0, stream>>>(f2w, SW, 512ull * 2048);
    gemm_bf16_k<true><<<dim3(E / 32, Mf / 16), 32, 0, stream>>>(
        SIM, 2048, SW, 2048, T2, E, 0, Mf, E, 2048, 1.f, f2b, 0, T2, E);
  }
  add_inplace<<<cdiv(2359296ull, 256), 256, 0, stream>>>(T2, XR, 2359296ull);

  // ------------------------- Decoder -------------------------
  // dec0: up 512->256 @48, concat S3, block 512->256
  { size_t t = 8ull * 512 * 48 * 48;
    upsample2x<<<cdiv(t, 256), 256, 0, stream>>>(T2, T1, 24, 24, t); }
  conv3x3(stream, T1, 8, 512, 48, 48, P(66), P(67), 256, CC, 0, 512, SIM, SW);
  { size_t t = 8ull * 256 * 2304;
    copy_ch<<<cdiv(t, 256), 256, 0, stream>>>(S3, CC, 256, 512, 256, 2304, t); }
  cbr(stream, CC, 8, 512, 48, 48, P(58), P(59), P(60), P(61), 256, T1, T2, MEAN, ISTD, SIM, SW);
  cbr(stream, T2, 8, 256, 48, 48, P(62), P(63), P(64), P(65), 256, T1, T2, MEAN, ISTD, SIM, SW);

  // dec1: up 256->128 @96, concat S2, block 256->128
  { size_t t = 8ull * 256 * 96 * 96;
    upsample2x<<<cdiv(t, 256), 256, 0, stream>>>(T2, T1, 48, 48, t); }
  conv3x3(stream, T1, 8, 256, 96, 96, P(76), P(77), 128, CC, 0, 256, SIM, SW);
  { size_t t = 8ull * 128 * 9216;
    copy_ch<<<cdiv(t, 256), 256, 0, stream>>>(S2, CC, 128, 256, 128, 9216, t); }
  cbr(stream, CC, 8, 256, 96, 96, P(68), P(69), P(70), P(71), 128, T1, T2, MEAN, ISTD, SIM, SW);
  cbr(stream, T2, 8, 128, 96, 96, P(72), P(73), P(74), P(75), 128, T1, T2, MEAN, ISTD, SIM, SW);

  // dec2: up 128->64 @192, concat S1, block 128->64
  { size_t t = 8ull * 128 * 192 * 192;
    upsample2x<<<cdiv(t, 256), 256, 0, stream>>>(T2, T1, 96, 96, t); }
  conv3x3(stream, T1, 8, 128, 192, 192, P(86), P(87), 64, CC, 0, 128, SIM, SW);
  { size_t t = 8ull * 64 * 36864;
    copy_ch<<<cdiv(t, 256), 256, 0, stream>>>(S1, CC, 64, 128, 64, 36864, t); }
  cbr(stream, CC, 8, 128, 192, 192, P(78), P(79), P(80), P(81), 64, T1, T2, MEAN, ISTD, SIM, SW);
  cbr(stream, T2, 8,  64, 192, 192, P(82), P(83), P(84), P(85), 64, T1, T2, MEAN, ISTD, SIM, SW);

  // final 1x1 conv 64 -> 1
  { size_t t = 8ull * 36864;
    conv1x1_64_1<<<cdiv(t, 256), 256, 0, stream>>>(T2, P(88), P(89), (float*)d_out, 36864, t); }
}
